// VideoMusicTransformer_V1_24489903522347
// MI455X (gfx1250) — matrix operand order repaired and checked by reading, not compiled, not run on hardware
//
#include <hip/hip_runtime.h>

// ============================================================================
// VideoMusicTransformer forward for MI455X (gfx1250, wave32, WMMA).
//
// - All GEMMs: v_wmma_f32_16x16x32_f16 (f16 in, f32 accumulate). The 64x64
//   B (weight) tile per K-stage is staged into LDS with double-buffered
//   global_load_async_to_lds_b128 (ASYNCcnt protocol: s_wait_asynccnt /
//   s_wait_dscnt + workgroup barriers), 8 WMMAs per barrier pair.
// - Attention: per-wave flash kernel (online softmax, LDS P-transpose with
//   explicit s_wait_dscnt).
// - MoE: all 6 experts dense (like the reference einsum) weighted by the
//   top-2 softmax gate -> mathematically identical to sparse routing.
//
// Param leaves assumed flattened in jax.tree_util order (dict keys sorted
// ASCII). Leaf index map below.
// ============================================================================

typedef _Float16 h8   __attribute__((ext_vector_type(8)));
typedef _Float16 v16h __attribute__((ext_vector_type(16)));
typedef float    v8f  __attribute__((ext_vector_type(8)));

// ---- model constants -------------------------------------------------------
#define cB   32
#define cS   300
#define cT   (cB * cS)      // 9600 tokens per stream
#define cD   512
#define cH   8
#define cHD  64
#define cFF  1024
#define cE   6
#define cNC  159
#define cKVP 320            // padded key stride for V^T buffer
#define cVK  832            // padded K for the 776-wide visual projection

// ---- epilogue modes --------------------------------------------------------
enum { EPI_ST32 = 0, EPI_SILU32 = 1, EPI_MUL16 = 2, EPI_GATEACC = 3, EPI_ST16 = 4 };

// ---- device helpers --------------------------------------------------------
__device__ __forceinline__ v8f vzero8() {
  v8f z;
#pragma unroll
  for (int i = 0; i < 8; ++i) z[i] = 0.0f;
  return z;
}

// A fragment (16x32 f16): lane = row (mod 16); elements 0..7 <- K = k+8*half,
// elements 8..15 <- K = k+16+8*half. Caller passes p = rowbase + k0 + 8*half.
__device__ __forceinline__ v16h ld_afrag(const _Float16* p) {
  h8 c0 = *(const h8*)(p);
  h8 c1 = *(const h8*)(p + 16);
  v16h a;
#pragma unroll
  for (int i = 0; i < 8; ++i) { a[i] = c0[i]; a[i + 8] = c1[i]; }
  return a;
}

// B fragment (32x16 f16, B stored transposed [N][K]): lane = col (mod 16);
// elements 0..7 <- K = k0+8*half, 8..15 <- K = k0+16+8*half. Caller passes p = colbase + k0 + 8*half.
__device__ __forceinline__ v16h ld_bfrag(const _Float16* p) {
  h8 c0 = *(const h8*)(p);
  h8 c1 = *(const h8*)(p + 16);
  v16h b;
#pragma unroll
  for (int i = 0; i < 8; ++i) { b[i] = c0[i]; b[i + 8] = c1[i]; }
  return b;
}

__device__ __forceinline__ v8f wmma16(v16h a, v16h b, v8f c) {
  return __builtin_amdgcn_wmma_f32_16x16x32_f16(false, a, false, b, (short)0, c,
                                                false, false);
}

// LDS byte-offset of a __shared__ object (addrspace(3) pointer value).
typedef __attribute__((address_space(3))) const void* lds_cptr_t;
__device__ __forceinline__ unsigned lds_off_u32(const void* p) {
  return (unsigned)(unsigned long long)(lds_cptr_t)p;
}

// Per-lane async copy of 16 bytes global -> LDS (tracked by ASYNCcnt).
__device__ __forceinline__ void async_copy_b128(unsigned lds_byte_off,
                                                const _Float16* src) {
  asm volatile("global_load_async_to_lds_b128 %0, %1, off"
               :: "v"(lds_byte_off), "v"(src)
               : "memory");
}

// ============================================================================
// Generic WMMA GEMM: C[M,N] = epi(A[M,K] @ B[K,N]), B pre-transposed [N,K] f16.
// 8 waves per block: wave (wm,wn) in 4x2 grid, each owns 32x32 (2x2 WMMA).
// The 64x64 B tile per K-stage is staged into LDS with double-buffered
// global_load_async_to_lds_b128 (2 x b128 per thread per stage), giving
// 8 WMMAs per barrier pair and removing the 4x redundant B loads across the
// M-waves. Requires K % 64 == 0.
// ============================================================================
__global__ __launch_bounds__(256)
void k_gemm(const _Float16* __restrict__ A, int lda,
            const _Float16* __restrict__ Bt, int ldb,
            float* __restrict__ C, int ldc,
            _Float16* __restrict__ C16, int ldc16,
            int M, int N, int K,
            const float* __restrict__ bias,
            const float* __restrict__ aux, int ldaux,
            const float* __restrict__ rowscale, int rss,
            float alpha, int epi) {
  __shared__ __align__(16) _Float16 bsm[2][64 * 64];   // double-buffered B tile
  const int lane = threadIdx.x & 31;
  const int wave = threadIdx.x >> 5;
  const int wm = wave >> 1;          // 0..3
  const int wn = wave & 1;           // 0..1
  const int bm = blockIdx.x * 128 + wm * 32;
  const int bn0 = blockIdx.y * 64;
  const int bn = bn0 + wn * 32;
  const int hlf = lane >> 4;
  const int l15 = lane & 15;

  v8f acc[2][2];
  acc[0][0] = vzero8(); acc[0][1] = vzero8();
  acc[1][0] = vzero8(); acc[1][1] = vzero8();

  int rA0 = bm + l15;       rA0 = rA0 < M ? rA0 : M - 1;
  int rA1 = bm + 16 + l15;  rA1 = rA1 < M ? rA1 : M - 1;
  const _Float16* a0p = A + (size_t)rA0 * lda + hlf * 8;
  const _Float16* a1p = A + (size_t)rA1 * lda + hlf * 8;

  // async-copy mapping: thread i stages 2 x 8 halfs of column (i>>2)
  const int tcol = threadIdx.x >> 2;                 // 0..63 (local column)
  const int tch = threadIdx.x & 3;                   // 0..3  (16B chunk in K)
  int bcol = bn0 + tcol; bcol = bcol < N ? bcol : N - 1;
  const _Float16* bsrc = Bt + (size_t)bcol * ldb + tch * 8;
  const unsigned ldsoff =
      lds_off_u32(&bsm[0][0]) + (unsigned)(tcol * 64 + tch * 8) * 2u;

  const int nk = K >> 6;                             // 64-wide K stages
  async_copy_b128(ldsoff, bsrc);                     // prime stage 0 -> buf 0
  async_copy_b128(ldsoff + 64u, bsrc + 32);
  for (int it = 0; it < nk; ++it) {
    const int k0 = it << 6;
    const int cb = it & 1;
    if (it + 1 < nk) {
      const unsigned nb = (unsigned)(((it + 1) & 1) << 13);  // other buf (bytes)
      async_copy_b128(ldsoff + nb, bsrc + k0 + 64);
      async_copy_b128(ldsoff + nb + 64u, bsrc + k0 + 96);
      asm volatile("s_wait_asynccnt 0x2" ::: "memory");  // current stage landed
    } else {
      asm volatile("s_wait_asynccnt 0x0" ::: "memory");
    }
    __syncthreads();                                 // tile visible to all waves
    const _Float16* lbb = &bsm[cb][0] + (wn * 32 + l15) * 64 + hlf * 8;
#pragma unroll
    for (int ks = 0; ks < 2; ++ks) {                 // two 32-wide sub-steps
      v16h bf0 = ld_bfrag(lbb + ks * 32);            // ds_load_b128 x2
      v16h bf1 = ld_bfrag(lbb + 16 * 64 + ks * 32);
      v16h af0 = ld_afrag(a0p + k0 + ks * 32);
      v16h af1 = ld_afrag(a1p + k0 + ks * 32);
      acc[0][0] = wmma16(af0, bf0, acc[0][0]);
      acc[0][1] = wmma16(af0, bf1, acc[0][1]);
      acc[1][0] = wmma16(af1, bf0, acc[1][0]);
      acc[1][1] = wmma16(af1, bf1, acc[1][1]);
    }
    asm volatile("s_wait_dscnt 0x0" ::: "memory");   // frags read out of LDS
    __syncthreads();                                 // safe to overwrite buf
  }

#pragma unroll
  for (int mi = 0; mi < 2; ++mi)
#pragma unroll
    for (int ni = 0; ni < 2; ++ni)
#pragma unroll
      for (int r = 0; r < 8; ++r) {
        int m = bm + mi * 16 + r + hlf * 8;
        int n = bn + ni * 16 + l15;
        if (m < M && n < N) {
          float v = acc[mi][ni][r];
          if (bias) v += bias[n];
          v *= alpha;
          switch (epi) {
            case EPI_ST32:
              C[(size_t)m * ldc + n] = v; break;
            case EPI_SILU32:
              C[(size_t)m * ldc + n] = v / (1.0f + __expf(-v)); break;
            case EPI_MUL16:
              C16[(size_t)m * ldc16 + n] =
                  (_Float16)(v * aux[(size_t)m * ldaux + n]); break;
            case EPI_GATEACC:
              C[(size_t)m * ldc + n] += rowscale[(size_t)m * rss] * v; break;
            case EPI_ST16:
              C16[(size_t)m * ldc16 + n] = (_Float16)v; break;
          }
        }
      }
}

// ============================================================================
// Per-wave flash attention. Q,K: f16 [T,512] head-sliced; VT: f16 [B,512,320]
// (row = head-dim channel, contiguous keys). One wave = one 16-row Q tile.
// ============================================================================
__global__ __launch_bounds__(256)
void k_flash(const _Float16* __restrict__ Q, const _Float16* __restrict__ Kb,
             const _Float16* __restrict__ VT, _Float16* __restrict__ O,
             int Sq, int Skv, int causal) {
  __shared__ __align__(16) _Float16 pl[8][16][32];   // per-wave P staging
  const int lane = threadIdx.x & 31;
  const int wave = threadIdx.x >> 5;
  const int hlf = lane >> 4;
  const int l15 = lane & 15;
  const int nqt = (Sq + 15) / 16;
  const int qt = blockIdx.x * 8 + wave;
  const int hh = blockIdx.y;
  const int bb = blockIdx.z;
  if (qt >= nqt) return;
  const int qbase = qt * 16;

  int qrow = qbase + l15; if (qrow >= Sq) qrow = Sq - 1;
  const _Float16* qp = Q + ((size_t)(bb * Sq + qrow) * cD + hh * cHD);
  v16h qf0 = ld_afrag(qp + 0 + hlf * 8);
  v16h qf1 = ld_afrag(qp + 32 + hlf * 8);

  v8f o[4];
#pragma unroll
  for (int j = 0; j < 4; ++j) o[j] = vzero8();
  float mi[8], li[8];
#pragma unroll
  for (int r = 0; r < 8; ++r) { mi[r] = -1e30f; li[r] = 0.0f; }

  const int nkc = (Skv + 31) / 32;
  for (int kc = 0; kc < nkc; ++kc) {
    const int kb0 = kc * 32;
    v8f s[2]; s[0] = vzero8(); s[1] = vzero8();
#pragma unroll
    for (int g = 0; g < 2; ++g) {
      int key = kb0 + g * 16 + l15;
      int keyc = key < Skv ? key : Skv - 1;
      const _Float16* kp = Kb + ((size_t)(bb * Skv + keyc) * cD + hh * cHD);
      v16h kf0 = ld_bfrag(kp + 0 + hlf * 8);
      v16h kf1 = ld_bfrag(kp + 32 + hlf * 8);
      s[g] = wmma16(qf0, kf0, s[g]);
      s[g] = wmma16(qf1, kf1, s[g]);
    }
    // mask (pad keys hard-masked; causal matches reference's additive -1e9)
    float sc[2][8];
#pragma unroll
    for (int g = 0; g < 2; ++g) {
      int key = kb0 + g * 16 + l15;
#pragma unroll
      for (int r = 0; r < 8; ++r) {
        float v = s[g][r];
        int qr = qbase + r + hlf * 8;
        if (key >= Skv) v = -1e30f;
        else if (causal && key > qr) v += -1e9f;
        sc[g][r] = v;
      }
    }
    // online softmax per row (row lives across the 16 lanes of one half)
#pragma unroll
    for (int r = 0; r < 8; ++r) {
      float rm = fmaxf(sc[0][r], sc[1][r]);
#pragma unroll
      for (int off = 1; off < 16; off <<= 1) rm = fmaxf(rm, __shfl_xor(rm, off, 32));
      float mn = fmaxf(mi[r], rm);
      float scale = __expf(mi[r] - mn);
      float p0 = __expf(sc[0][r] - mn);
      float p1 = __expf(sc[1][r] - mn);
      float rs = p0 + p1;
#pragma unroll
      for (int off = 1; off < 16; off <<= 1) rs += __shfl_xor(rs, off, 32);
      li[r] = li[r] * scale + rs;
      mi[r] = mn;
#pragma unroll
      for (int j = 0; j < 4; ++j) o[j][r] *= scale;
      int mloc = r + hlf * 8;
      pl[wave][mloc][0 * 16 + l15] = (_Float16)p0;
      pl[wave][mloc][1 * 16 + l15] = (_Float16)p1;
    }
    // wave-private LDS: wait for DS stores, then read P back in A-layout
    asm volatile("s_wait_dscnt 0" ::: "memory");
    const _Float16* plp = &pl[wave][l15][hlf * 8];
    h8 c0 = *(const h8*)plp;
    h8 c1 = *(const h8*)(plp + 16);
    v16h pf;
#pragma unroll
    for (int i = 0; i < 8; ++i) { pf[i] = c0[i]; pf[i + 8] = c1[i]; }
#pragma unroll
    for (int j = 0; j < 4; ++j) {
      const _Float16* vp =
          VT + ((size_t)bb * cD + hh * cHD + j * 16 + l15) * cKVP + kb0 + hlf * 8;
      v16h vfrag = ld_bfrag(vp);
      o[j] = wmma16(pf, vfrag, o[j]);
    }
  }
  // normalize + store
#pragma unroll
  for (int r = 0; r < 8; ++r) {
    float inv = 1.0f / li[r];
    int row = qbase + r + hlf * 8;
    if (row < Sq) {
#pragma unroll
      for (int j = 0; j < 4; ++j)
        O[((size_t)(bb * Sq + row) * cD) + hh * cHD + j * 16 + l15] =
            (_Float16)(o[j][r] * inv);
    }
  }
}

// ============================================================================
// Small elementwise / reduction kernels
// ============================================================================
__global__ void k_convT(const float* __restrict__ src, _Float16* __restrict__ dst,
                        int K, int N, int Kpad) {
  int i = blockIdx.x * 256 + threadIdx.x;
  if (i >= N * Kpad) return;
  int n = i / Kpad, k = i - n * Kpad;
  dst[i] = (k < K) ? (_Float16)src[(size_t)k * N + n] : (_Float16)0.0f;
}

__global__ void k_embed(const int* __restrict__ xr, const int* __restrict__ xa,
                        const float* __restrict__ er, const float* __restrict__ ea,
                        _Float16* __restrict__ xe) {
  int i = blockIdx.x * 256 + threadIdx.x;
  if (i >= cT * cD) return;
  int t = i >> 9, d = i & 511;
  xe[i] = (_Float16)(er[xr[t] * cD + d] + ea[xa[t] * cD + d]);
}

__global__ void k_packvis(const float* __restrict__ sem, const float* __restrict__ sce,
                          const float* __restrict__ mot, const float* __restrict__ emo,
                          _Float16* __restrict__ dst) {
  int i = blockIdx.x * 256 + threadIdx.x;
  if (i >= cT * cVK) return;
  int t = i / cVK, c = i - t * cVK;
  float v;
  if (c < 768)      v = sem[(size_t)t * 768 + c];
  else if (c == 768) v = sce[t];
  else if (c == 769) v = mot[t];
  else if (c < 776)  v = emo[t * 6 + (c - 770)];
  else               v = 0.0f;
  dst[i] = (_Float16)v;
}

__global__ void k_chordfin(const float* __restrict__ acc, const float* __restrict__ wlast,
                           const float* __restrict__ key, const float* __restrict__ pos,
                           float* __restrict__ y, _Float16* __restrict__ y16) {
  int i = blockIdx.x * 256 + threadIdx.x;
  if (i >= cT * cD) return;
  int t = i >> 9, d = i & 511;
  int b = t / cS, s = t - b * cS;
  float v = acc[i] + key[b] * wlast[d] + pos[(size_t)s * cD + d];
  y[i] = v; y16[i] = (_Float16)v;
}

__global__ void k_visfin(const float* __restrict__ acc, const float* __restrict__ pos,
                         float* __restrict__ y, _Float16* __restrict__ y16) {
  int i = blockIdx.x * 256 + threadIdx.x;
  if (i >= cT * cD) return;
  int t = i >> 9, d = i & 511;
  int s = t % cS;
  float v = acc[i] + pos[(size_t)s * cD + d];
  y[i] = v; y16[i] = (_Float16)v;
}

__global__ void k_vt(const float* __restrict__ V, _Float16* __restrict__ VT) {
  int i = blockIdx.x * 256 + threadIdx.x;
  if (i >= cB * cD * cS) return;
  int s = i % cS; int rest = i / cS;
  int c = rest % cD; int b = rest / cD;
  VT[((size_t)b * cD + c) * cKVP + s] = (_Float16)V[((size_t)(b * cS + s)) * cD + c];
}

__global__ __launch_bounds__(128)
void k_ln(const float* __restrict__ x, const float* __restrict__ f,
          const float* __restrict__ g, const float* __restrict__ b,
          float* __restrict__ y, _Float16* __restrict__ y16) {
  __shared__ float red[4];
  const int t = blockIdx.x, tid = threadIdx.x;
  const float* xp = x + (size_t)t * cD;
  float v[4];
#pragma unroll
  for (int i = 0; i < 4; ++i) {
    int d = tid + i * 128;
    float a = xp[d];
    if (f) a += f[(size_t)t * cD + d];
    v[i] = a;
  }
  float s = v[0] + v[1] + v[2] + v[3];
#pragma unroll
  for (int off = 16; off > 0; off >>= 1) s += __shfl_xor(s, off, 32);
  if ((tid & 31) == 0) red[tid >> 5] = s;
  __syncthreads();
  float mean = (red[0] + red[1] + red[2] + red[3]) * (1.0f / cD);
  __syncthreads();
  float q = 0.0f;
#pragma unroll
  for (int i = 0; i < 4; ++i) { float d2 = v[i] - mean; q += d2 * d2; }
#pragma unroll
  for (int off = 16; off > 0; off >>= 1) q += __shfl_xor(q, off, 32);
  if ((tid & 31) == 0) red[tid >> 5] = q;
  __syncthreads();
  float inv = rsqrtf((red[0] + red[1] + red[2] + red[3]) * (1.0f / cD) + 1e-5f);
#pragma unroll
  for (int i = 0; i < 4; ++i) {
    int d = tid + i * 128;
    float o = (v[i] - mean) * inv * g[d] + b[d];
    y[(size_t)t * cD + d] = o;
    y16[(size_t)t * cD + d] = (_Float16)o;
  }
}

__global__ void k_gate(const float* __restrict__ x, const float* __restrict__ Wg,
                       const float* __restrict__ bg, float* __restrict__ logits) {
  int i = blockIdx.x * 256 + threadIdx.x;
  if (i >= cT * cE) return;
  int t = i / cE, e = i - t * cE;
  float s = bg[e];
  const float* xp = x + (size_t)t * cD;
  for (int k = 0; k < cD; ++k) s += xp[k] * Wg[k * cE + e];
  logits[i] = s;
}

__global__ void k_top2(const float* __restrict__ logits, float* __restrict__ gw) {
  int t = blockIdx.x * 256 + threadIdx.x;
  if (t >= cT) return;
  float l[cE];
#pragma unroll
  for (int e = 0; e < cE; ++e) l[e] = logits[t * cE + e];
  int i1 = 0;
#pragma unroll
  for (int e = 1; e < cE; ++e) if (l[e] > l[i1]) i1 = e;   // first max (jax tie rule)
  int i2 = -1;
#pragma unroll
  for (int e = 0; e < cE; ++e) {
    if (e == i1) continue;
    if (i2 < 0 || l[e] > l[i2]) i2 = e;
  }
  float e2 = __expf(l[i2] - l[i1]);
  float w1 = 1.0f / (1.0f + e2);
#pragma unroll
  for (int e = 0; e < cE; ++e) gw[t * cE + e] = 0.0f;
  gw[t * cE + i1] = w1;
  gw[t * cE + i2] = 1.0f - w1;
}

// ============================================================================
// Host orchestration
// ============================================================================
struct Ws {
  float *mem32, *x32, *f32a, *t1, *logits, *gatew;
  _Float16 *mem16, *x16, *q16, *k16, *ao16, *vt16, *h16, *slot;
};

// --- param leaf index map (jax.tree_util sorted-key flattening) -------------
// 0..7 : x, x_root, x_attr, feature_semantic, feature_key, feature_scene_offset,
//        feature_motion, feature_emotion
#define P0         8
#define P_WCHORD   (P0 + 0)
#define P_WVIS     (P0 + 1)
#define P_WOUT     (P0 + 2)
#define P_BCHORD   (P0 + 3)
#define P_BVIS     (P0 + 4)
#define P_BOUT     (P0 + 5)
#define P_DEC      (P0 + 6)          // 6 layers x 30 leaves
#define DEC_STRIDE 30
#define DL_CA      0                 // Wk,Wo,Wq,Wv,bk,bo,bq,bv
#define DL_LN1     8                 // b,g
#define DL_LN2     10
#define DL_LN3     12
#define DL_MOE     14                // W1,W2,W3,Wg,b1,b2,b3,bg
#define DL_SA      22
#define P_DECNORM  (P_DEC + 6 * DEC_STRIDE)   // b,g
#define P_EMBATTR  (P_DECNORM + 2)
#define P_EMBROOT  (P_EMBATTR + 1)
#define P_ENC      (P_EMBROOT + 1)   // 6 layers x 22 leaves
#define ENC_STRIDE 22
#define EL_LN1     0
#define EL_LN2     2
#define EL_MOE     4
#define EL_SA      12
#define P_ENCNORM  (P_ENC + 6 * ENC_STRIDE)
#define P_POS      (P_ENCNORM + 2)
#define P_POSV     (P_POS + 1)
// attn leaves: Wk=0 Wo=1 Wq=2 Wv=3 bk=4 bo=5 bq=6 bv=7
// moe  leaves: W1=0 W2=1 W3=2 Wg=3 b1=4 b2=5 b3=6 bg=7

static inline void convT(hipStream_t st, const float* src, _Float16* dst,
                         int K, int N, int Kpad) {
  int n = N * Kpad;
  k_convT<<<(n + 255) / 256, 256, 0, st>>>(src, dst, K, N, Kpad);
}

static inline void gemm(hipStream_t st, const _Float16* A, int lda,
                        const _Float16* Bt, int ldb, float* C, int ldc,
                        _Float16* C16, int ldc16, int M, int N, int K,
                        const float* bias, const float* aux, int ldaux,
                        const float* rowsc, int rss, float alpha, int epi) {
  dim3 g((M + 127) / 128, (N + 63) / 64);
  k_gemm<<<g, 256, 0, st>>>(A, lda, Bt, ldb, C, ldc, C16, ldc16, M, N, K,
                            bias, aux, ldaux, rowsc, rss, alpha, epi);
}

static void attn_block(hipStream_t st, Ws& w, const float* const* P, int ab,
                       const _Float16* xq16, const _Float16* xkv16, int causal) {
  // Q (scale 1/sqrt(64) folded in)
  convT(st, P[ab + 2], w.slot, cD, cD, cD);
  gemm(st, xq16, cD, w.slot, cD, nullptr, 0, w.q16, cD, cT, cD, cD,
       P[ab + 6], nullptr, 0, nullptr, 0, 0.125f, EPI_ST16);
  // K
  convT(st, P[ab + 0], w.slot, cD, cD, cD);
  gemm(st, xkv16, cD, w.slot, cD, nullptr, 0, w.k16, cD, cT, cD, cD,
       P[ab + 4], nullptr, 0, nullptr, 0, 1.0f, EPI_ST16);
  // V (f32) then transpose to [B,512,320] f16
  convT(st, P[ab + 3], w.slot, cD, cD, cD);
  gemm(st, xkv16, cD, w.slot, cD, w.f32a, cD, nullptr, 0, cT, cD, cD,
       P[ab + 7], nullptr, 0, nullptr, 0, 1.0f, EPI_ST32);
  k_vt<<<(cB * cD * cS + 255) / 256, 256, 0, st>>>(w.f32a, w.vt16);
  // flash attention -> ao16
  dim3 fg(((cS + 15) / 16 + 7) / 8, cH, cB);
  k_flash<<<fg, 256, 0, st>>>(w.q16, w.k16, w.vt16, w.ao16, cS, cS, causal);
  // output projection -> f32a
  convT(st, P[ab + 1], w.slot, cD, cD, cD);
  gemm(st, w.ao16, cD, w.slot, cD, w.f32a, cD, nullptr, 0, cT, cD, cD,
       P[ab + 5], nullptr, 0, nullptr, 0, 1.0f, EPI_ST32);
}

static void moe_block(hipStream_t st, Ws& w, const float* const* P, int mb) {
  k_gate<<<(cT * cE + 255) / 256, 256, 0, st>>>(w.x32, P[mb + 3], P[mb + 7], w.logits);
  k_top2<<<(cT + 255) / 256, 256, 0, st>>>(w.logits, w.gatew);
  hipMemsetAsync(w.f32a, 0, (size_t)cT * cD * sizeof(float), st);
  for (int e = 0; e < cE; ++e) {
    // t1 = silu(x @ W1e + b1e)
    convT(st, P[mb + 0] + (size_t)e * cD * cFF, w.slot, cD, cFF, cD);
    gemm(st, w.x16, cD, w.slot, cD, w.t1, cFF, nullptr, 0, cT, cFF, cD,
         P[mb + 4] + e * cFF, nullptr, 0, nullptr, 0, 1.0f, EPI_SILU32);
    // h16 = (x @ W2e + b2e) * t1
    convT(st, P[mb + 1] + (size_t)e * cD * cFF, w.slot, cD, cFF, cD);
    gemm(st, w.x16, cD, w.slot, cD, nullptr, 0, w.h16, cFF, cT, cFF, cD,
         P[mb + 5] + e * cFF, w.t1, cFF, nullptr, 0, 1.0f, EPI_MUL16);
    // f32a += gate[t,e] * (h16 @ W3e + b3e)
    convT(st, P[mb + 2] + (size_t)e * cFF * cD, w.slot, cFF, cD, cFF);
    gemm(st, w.h16, cFF, w.slot, cFF, w.f32a, cD, nullptr, 0, cT, cD, cFF,
         P[mb + 6] + e * cD, nullptr, 0, w.gatew + e, cE, 1.0f, EPI_GATEACC);
  }
}

static inline void ln(hipStream_t st, const float* x, const float* f,
                      const float* b, const float* g, float* y, _Float16* y16) {
  k_ln<<<cT, 128, 0, st>>>(x, f, g, b, y, y16);
}

extern "C" void kernel_launch(void* const* d_in, const int* in_sizes, int n_in,
                              void* d_out, int out_size, void* d_ws, size_t ws_size,
                              hipStream_t stream) {
  (void)in_sizes; (void)n_in; (void)out_size; (void)ws_size;
  const float* const* P = (const float* const*)d_in;
  const int* x_root = (const int*)d_in[1];
  const int* x_attr = (const int*)d_in[2];
  const float* f_sem = (const float*)d_in[3];
  const float* f_key = (const float*)d_in[4];
  const float* f_sce = (const float*)d_in[5];
  const float* f_mot = (const float*)d_in[6];
  const float* f_emo = (const float*)d_in[7];

  // ---- workspace carve ----
  char* base = (char*)d_ws;
  auto carve = [&](size_t bytes) -> void* {
    void* p = (void*)base;
    base += (bytes + 255) & ~(size_t)255;
    return p;
  };
  Ws w;
  w.mem32  = (float*)carve((size_t)cT * cD * 4);
  w.x32    = (float*)carve((size_t)cT * cD * 4);
  w.f32a   = (float*)carve((size_t)cT * cD * 4);
  w.t1     = (float*)carve((size_t)cT * cFF * 4);
  w.logits = (float*)carve((size_t)cT * cE * 4);
  w.gatew  = (float*)carve((size_t)cT * cE * 4);
  w.mem16  = (_Float16*)carve((size_t)cT * cD * 2);
  w.x16    = (_Float16*)carve((size_t)cT * cD * 2);
  w.q16    = (_Float16*)carve((size_t)cT * cD * 2);
  w.k16    = (_Float16*)carve((size_t)cT * cD * 2);
  w.ao16   = (_Float16*)carve((size_t)cT * cD * 2);
  w.vt16   = (_Float16*)carve((size_t)cB * cD * cKVP * 2);
  w.h16    = (_Float16*)carve((size_t)cT * cFF * 2);
  w.slot   = (_Float16*)carve((size_t)cFF * cD * 2);   // weight conversion slot

  hipMemsetAsync(w.vt16, 0, (size_t)cB * cD * cKVP * 2, stream);

  // ---- visual feature projection -> encoder stream (x32/x16) ----
  k_packvis<<<(cT * cVK + 255) / 256, 256, 0, stream>>>(f_sem, f_sce, f_mot, f_emo, w.h16);
  convT(stream, P[P_WVIS], w.slot, 776, cD, cVK);
  gemm(stream, w.h16, cVK, w.slot, cVK, w.f32a, cD, nullptr, 0, cT, cD, cVK,
       P[P_BVIS], nullptr, 0, nullptr, 0, 1.0f, EPI_ST32);
  k_visfin<<<(cT * cD + 255) / 256, 256, 0, stream>>>(w.f32a, P[P_POSV], w.x32, w.x16);

  // ---- encoder ----
  for (int l = 0; l < 6; ++l) {
    int lb = P_ENC + l * ENC_STRIDE;
    attn_block(stream, w, P, lb + EL_SA, w.x16, w.x16, 0);
    ln(stream, w.x32, w.f32a, P[lb + EL_LN1 + 0], P[lb + EL_LN1 + 1], w.x32, w.x16);
    moe_block(stream, w, P, lb + EL_MOE);
    ln(stream, w.x32, w.f32a, P[lb + EL_LN2 + 0], P[lb + EL_LN2 + 1], w.x32, w.x16);
  }
  ln(stream, w.x32, nullptr, P[P_ENCNORM + 0], P[P_ENCNORM + 1], w.mem32, w.mem16);

  // ---- chord embedding -> decoder stream (x32/x16) ----
  k_embed<<<(cT * cD + 255) / 256, 256, 0, stream>>>(x_root, x_attr,
                                                     P[P_EMBROOT], P[P_EMBATTR], w.ao16);
  convT(stream, P[P_WCHORD], w.slot, cD, cD, cD);   // first 512 rows of [513,512]
  gemm(stream, w.ao16, cD, w.slot, cD, w.f32a, cD, nullptr, 0, cT, cD, cD,
       P[P_BCHORD], nullptr, 0, nullptr, 0, 1.0f, EPI_ST32);
  k_chordfin<<<(cT * cD + 255) / 256, 256, 0, stream>>>(
      w.f32a, P[P_WCHORD] + (size_t)cD * cD, f_key, P[P_POS], w.x32, w.x16);

  // ---- decoder ----
  for (int l = 0; l < 6; ++l) {
    int lb = P_DEC + l * DEC_STRIDE;
    attn_block(stream, w, P, lb + DL_SA, w.x16, w.x16, 1);      // causal self-attn
    ln(stream, w.x32, w.f32a, P[lb + DL_LN1 + 0], P[lb + DL_LN1 + 1], w.x32, w.x16);
    attn_block(stream, w, P, lb + DL_CA, w.x16, w.mem16, 0);    // cross-attn
    ln(stream, w.x32, w.f32a, P[lb + DL_LN2 + 0], P[lb + DL_LN2 + 1], w.x32, w.x16);
    moe_block(stream, w, P, lb + DL_MOE);
    ln(stream, w.x32, w.f32a, P[lb + DL_LN3 + 0], P[lb + DL_LN3 + 1], w.x32, w.x16);
  }
  ln(stream, w.x32, nullptr, P[P_DECNORM + 0], P[P_DECNORM + 1], w.x32, w.x16);

  // ---- output head: [T,512] @ [512,159] + bout -> d_out f32 ----
  convT(stream, P[P_WOUT], w.slot, cD, cNC, cD);
  gemm(stream, w.x16, cD, w.slot, cD, (float*)d_out, cNC, nullptr, 0,
       cT, cNC, cD, P[P_BOUT], nullptr, 0, nullptr, 0, 1.0f, EPI_ST32);
}
